// TransformerLayer_31447750542222
// MI455X (gfx1250) — matrix operand order repaired and checked by reading, not compiled, not run on hardware
//
#include <hip/hip_runtime.h>

// ---------------------------------------------------------------------------
// Types for CDNA5 WMMA (gfx1250, wave32)
// ---------------------------------------------------------------------------
typedef unsigned short u16;
typedef __attribute__((ext_vector_type(16))) __bf16       bf16x16;
typedef __attribute__((ext_vector_type(8)))  float        floatx8;
typedef __attribute__((ext_vector_type(4)))  unsigned int uint32x4;
typedef __attribute__((ext_vector_type(4)))  int          i32x4;
typedef __attribute__((ext_vector_type(8)))  int          i32x8;

union Frag {
  uint32x4 u[2];
  bf16x16  v;
  u16      s[16];
};

__device__ __forceinline__ u16 f2bf(float f) {
  unsigned int u = __float_as_uint(f);
  u += 0x7FFFu + ((u >> 16) & 1u);  // round-to-nearest-even
  return (u16)(u >> 16);
}

__device__ __forceinline__ floatx8 wmma_bf16(bf16x16 a, bf16x16 b, floatx8 c) {
  // v_wmma_f32_16x16x32_bf16: D = A(16x32) * B(32x16) + C(16x16 f32)
  return __builtin_amdgcn_wmma_f32_16x16x32_bf16(false, a, false, b, (short)0, c,
                                                 false, false);
}

// ---------------------------------------------------------------------------
// CDNA5 async global->LDS copy (ASYNCcnt path), with sync fallback.
// ---------------------------------------------------------------------------
#if defined(__AMDGCN__) && __has_builtin(__builtin_amdgcn_global_load_async_to_lds_b128)
#define USE_ASYNC 1
typedef __attribute__((address_space(1))) i32x4 gbl_i32x4;
typedef __attribute__((address_space(3))) i32x4 lds_i32x4;
#else
#define USE_ASYNC 0
#endif

__device__ __forceinline__ void async_copy16(u16* l, const u16* g) {
#if USE_ASYNC
  __builtin_amdgcn_global_load_async_to_lds_b128((gbl_i32x4*)g, (lds_i32x4*)l, 0, 0);
#else
  *(uint32x4*)l = *(const uint32x4*)g;
#endif
}

template <int N>
__device__ __forceinline__ void async_wait() {
#if USE_ASYNC
#if __has_builtin(__builtin_amdgcn_s_wait_asynccnt)
  __builtin_amdgcn_s_wait_asynccnt(N);
#else
  asm volatile("s_wait_asynccnt %0" ::"i"(N) : "memory");
#endif
#endif
}

// ---------------------------------------------------------------------------
// CDNA5 Tensor Data Mover: 2-D tile load global->LDS (TENSORcnt path).
// D# layout per CDNA5 ISA. Groups 2/3 zero (2-D tensor).
// This toolchain's builtin is the 6-arg form:
//   (uint32x4 g0, int32x8 g1, int32x4, int32x4, int32x8, i32 cpol)
// ---------------------------------------------------------------------------
#if defined(__AMDGCN__) && __has_builtin(__builtin_amdgcn_tensor_load_to_lds)
#define USE_TDM 1
__device__ __forceinline__ void tdm_load_2d(const u16* gsrc, unsigned ldsAddr,
                                            int tileRows /*tile_dim1*/,
                                            int tileCols /*tile_dim0*/,
                                            unsigned tensorW, unsigned tensorH,
                                            unsigned long long strideElems) {
  unsigned long long ga = (unsigned long long)(size_t)gsrc;
  uint32x4 g0;
  g0[0] = 1u;                                                   // count=1, user D#
  g0[1] = ldsAddr;                                              // LDS byte address
  g0[2] = (unsigned)ga;                                         // global_addr[31:0]
  g0[3] = (unsigned)((ga >> 32) & 0x01FFFFFFull) | (2u << 30);  // addr[56:32] | type=2
  i32x8 g1;
  g1[0] = 1 << 16;                                              // data_size=2 bytes
  g1[1] = (int)((tensorW & 0xFFFFu) << 16);                     // tensor_dim0 lo
  g1[2] = (int)((tensorW >> 16) | ((tensorH & 0xFFFFu) << 16)); // dim0 hi | dim1 lo
  g1[3] = (int)((tensorH >> 16) | ((unsigned)tileCols << 16));  // dim1 hi | tile_dim0
  g1[4] = tileRows & 0xFFFF;                                    // tile_dim1 (tile_dim2=0)
  g1[5] = (int)(unsigned)(strideElems & 0xFFFFFFFFull);         // dim0_stride lo
  g1[6] = (int)(unsigned)((strideElems >> 32) & 0xFFFFull);     // dim0_stride hi
  g1[7] = 0;
  i32x4 zz = (i32x4)0;
  i32x8 z8 = (i32x8)0;
  __builtin_amdgcn_tensor_load_to_lds(g0, g1, zz, zz, z8, 0);
}
#else
#define USE_TDM 0
#endif

// Problem constants (match reference)
#define BB   2
#define LL   2048
#define LLC  512
#define DM   1024
#define DC   768
#define DCND 256
#define NH   16
#define DH   64
#define DFF  3072

// ---------------------------------------------------------------------------
// Elementwise helpers
// ---------------------------------------------------------------------------
__global__ void copy_f32_kernel(const float* __restrict__ src, float* __restrict__ dst, int n) {
  int i = blockIdx.x * blockDim.x + threadIdx.x;
  if (i < n) dst[i] = src[i];
}

__global__ void cvt_bf16_kernel(const float* __restrict__ src, u16* __restrict__ dst, int n) {
  int i = blockIdx.x * blockDim.x + threadIdx.x;
  if (i < n) dst[i] = f2bf(src[i]);
}

// svec[b,d] = dot(cond[b,:], w[d,:]) + 1
__global__ void cond_proj_kernel(const float* __restrict__ cond, const float* __restrict__ w,
                                 float* __restrict__ svec, int D) {
  int i = blockIdx.x * blockDim.x + threadIdx.x;
  int b = i / D, d = i - b * D;
  const float* c = cond + (size_t)b * DCND;
  const float* wr = w + (size_t)d * DCND;
  float acc = 0.f;
  #pragma unroll 4
  for (int k = 0; k < DCND; ++k) acc += c[k] * wr[k];
  svec[i] = acc + 1.0f;
}

// out[row,d] = x[row,d] * svec[(b?),d] * rsqrt(mean(x^2)+eps)   (bf16 out)
__global__ __launch_bounds__(256) void ada_rms_kernel(const float* __restrict__ x,
                                                      const float* __restrict__ svec,
                                                      u16* __restrict__ out, int D,
                                                      int perBatch, int rowsPerBatch) {
  __shared__ float red[8];
  int row = blockIdx.x;
  const float* xr = x + (size_t)row * D;
  float ss = 0.f;
  for (int d = threadIdx.x; d < D; d += blockDim.x) { float v = xr[d]; ss += v * v; }
  #pragma unroll
  for (int m = 16; m >= 1; m >>= 1) ss += __shfl_xor(ss, m);
  if ((threadIdx.x & 31) == 0) red[threadIdx.x >> 5] = ss;
  __syncthreads();
  float tot = 0.f;
  #pragma unroll
  for (int i = 0; i < 8; ++i) tot += red[i];
  float inv = rsqrtf(tot / (float)D + 1e-6f);
  int b = row / rowsPerBatch;
  const float* sv = svec + (perBatch ? (size_t)b * D : 0);
  u16* o = out + (size_t)row * D;
  for (int d = threadIdx.x; d < D; d += blockDim.x) o[d] = f2bf(xr[d] * sv[d] * inv);
}

// t = a * silu(g), bf16 out.  u layout: [rows, 2*DFF]
__global__ void silu_mul_kernel(const float* __restrict__ u, u16* __restrict__ t, int n) {
  int i = blockIdx.x * blockDim.x + threadIdx.x;
  if (i >= n) return;
  int row = i / DFF, j = i - row * DFF;
  const float* ur = u + (size_t)row * (2 * DFF);
  float a = ur[j], g = ur[DFF + j];
  t[i] = f2bf(a * (g / (1.0f + __expf(-g))));
}

// ---------------------------------------------------------------------------
// Cosine-scale + 3D RoPE prep: src (per-row fp32, stride) -> dst bf16 [b,h,l,64]
// One wave per (b,h,l) row: lane owns elements e=lane and e=lane+32.
// ---------------------------------------------------------------------------
__global__ __launch_bounds__(128) void prep_qk_kernel(const float* __restrict__ src,
                                                      int rowStride, int colBase,
                                                      const float* __restrict__ pos,
                                                      const float* __restrict__ scale,
                                                      u16* __restrict__ dst, int Lrow) {
  int wid = (blockIdx.x * blockDim.x + threadIdx.x) >> 5;
  int lane = threadIdx.x & 31;
  int l = wid % Lrow;
  int bh = wid / Lrow;
  int h = bh % NH, b = bh / NH;

  const float* p = src + ((size_t)b * Lrow + l) * rowStride + colBase + h * DH;
  float v0 = p[lane], v1 = p[lane + 32];

  float ss = v0 * v0 + v1 * v1;
  #pragma unroll
  for (int m = 16; m >= 1; m >>= 1) ss += __shfl_xor(ss, m);
  float s = sqrtf(scale[h]) * rsqrtf(ss + 1e-6f);
  v0 *= s; v1 *= s;

  // uniform cross-lane fetch of rotation partners (x[e+-30])
  int pj0 = (lane < 30) ? (lane + 30) : (lane - 30);   // partner of e0 = lane
  int pj1 = (lane < 28) ? (lane + 2)  : (lane + 32);   // partner of e1 = lane+32
  float a0 = __shfl(v0, pj0 & 31), b0 = __shfl(v1, pj0 & 31);
  float a1 = __shfl(v0, pj1 & 31), b1 = __shfl(v1, pj1 & 31);
  float partner0 = (pj0 < 32) ? a0 : b0;
  float partner1 = (pj1 < 32) ? a1 : b1;

  const float* pr = pos + ((size_t)b * Lrow + l) * 3;
  float pv[3] = { pr[0], pr[1], pr[2] };
  const float LOGPI = 1.1447298858494002f;
  const float LSTEP = 2.302585092994046f / 160.0f;  // log(10)/160

  // e0 = lane (always < 60): d = e<30 ? e : e-30
  int d0 = (lane < 30) ? lane : (lane - 30);
  float th0 = pv[d0 / 10] * __expf(LOGPI + (float)((d0 % 10) * NH + h) * LSTEP);
  float c0 = __cosf(th0), s0 = __sinf(th0);
  float out0 = (lane < 30) ? (v0 * c0 - partner0 * s0) : (v0 * c0 + partner0 * s0);

  // e1 = lane+32: rotated (x2 branch) iff e1 < 60, i.e. lane < 28
  float out1 = v1;
  if (lane < 28) {
    int d1 = lane + 2;
    float th1 = pv[d1 / 10] * __expf(LOGPI + (float)((d1 % 10) * NH + h) * LSTEP);
    out1 = v1 * __cosf(th1) + partner1 * __sinf(th1);
  }

  u16* o = dst + (((size_t)bh * Lrow + l) * DH);
  o[lane] = f2bf(out0);
  o[lane + 32] = f2bf(out1);
}

__global__ void prep_v_kernel(const float* __restrict__ src, int rowStride, int colBase,
                              u16* __restrict__ dst, int Lrow, int n) {
  int i = blockIdx.x * blockDim.x + threadIdx.x;
  if (i >= n) return;
  int e = i & 63;
  int t = i >> 6;
  int l = t % Lrow;
  int t2 = t / Lrow;
  int h = t2 % NH, b = t2 / NH;
  dst[i] = f2bf(src[((size_t)b * Lrow + l) * rowStride + colBase + h * DH + e]);
}

// ---------------------------------------------------------------------------
// Tiled bf16 WMMA GEMM: C[M,N] (+)= A[M,K] * W[N,K]^T
// 128 threads (4 waves), 64x64 tile, BK=32, each wave 2x2 WMMA tiles.
// Double-buffered staging: TDM tensor_load_to_lds (wave 0 issues 2 D# per
// tile, s_wait_tensorcnt) with async-to-LDS fallback. Last iteration peeled.
// ---------------------------------------------------------------------------
__global__ __launch_bounds__(128) void gemm_bf16_kernel(const u16* __restrict__ A,
                                                        const u16* __restrict__ W,
                                                        float* __restrict__ C,
                                                        int M, int N, int K, int acc) {
  __shared__ alignas(16) u16 As[2][64 * 32];
  __shared__ alignas(16) u16 Bs[2][64 * 32];

  int tid = threadIdx.x;
  int lane = tid & 31, wid = tid >> 5;
  int hi = lane >> 4, ln = lane & 15;
  int m0 = blockIdx.y * 64, n0 = blockIdx.x * 64;
  int wy = wid >> 1, wx = wid & 1;

  floatx8 accv[2][2];
  #pragma unroll
  for (int i = 0; i < 2; ++i)
    #pragma unroll
    for (int j = 0; j < 2; ++j) accv[i][j] = (floatx8)0.0f;

  // stage one 64x32 K-slab of A and W into LDS buffer `buf`
  auto stage = [&](int buf, int kt) {
#if USE_TDM
    if (wid == 0) {
      tdm_load_2d(A + (size_t)m0 * K + (size_t)kt * 32, (unsigned)(size_t)&As[buf][0],
                  64, 32, (unsigned)K, (unsigned)(M - m0), (unsigned long long)K);
      tdm_load_2d(W + (size_t)n0 * K + (size_t)kt * 32, (unsigned)(size_t)&Bs[buf][0],
                  64, 32, (unsigned)K, (unsigned)(N - n0), (unsigned long long)K);
    }
#else
    int kb = kt * 32;
    #pragma unroll
    for (int p = 0; p < 2; ++p) {
      int idx = (p * 128 + tid) * 8;   // element offset in 64x32 tile
      int r = idx >> 5, c = idx & 31;
      async_copy16(&As[buf][idx], A + (size_t)(m0 + r) * K + kb + c);
      async_copy16(&Bs[buf][idx], W + (size_t)(n0 + r) * K + kb + c);
    }
#endif
  };

  auto compute = [&](int buf) {
    Frag af[2], bf[2];
    #pragma unroll
    for (int i = 0; i < 2; ++i) {
      int ra = (wy * 32 + i * 16 + ln) * 32;
      af[i].u[0] = *(const uint32x4*)(&As[buf][ra + hi * 8]);       // A: K = hi*8..+7
      af[i].u[1] = *(const uint32x4*)(&As[buf][ra + 16 + hi * 8]);  //    K = 16+hi*8..+7
      int rb = (wx * 32 + i * 16 + ln) * 32;
      bf[i].u[0] = *(const uint32x4*)(&Bs[buf][rb + hi * 8]);       // B: K = hi*8..+7
      bf[i].u[1] = *(const uint32x4*)(&Bs[buf][rb + 16 + hi * 8]);  //    K = 16+hi*8..+7
    }
    #pragma unroll
    for (int i = 0; i < 2; ++i)
      #pragma unroll
      for (int j = 0; j < 2; ++j)
        accv[i][j] = wmma_bf16(af[i].v, bf[j].v, accv[i][j]);
  };

  int nk = K >> 5;
  stage(0, 0);
  for (int kt = 0; kt + 1 < nk; ++kt) {     // steady state: no conditionals
    stage((kt + 1) & 1, kt + 1);
#if USE_TDM
    if (wid == 0) __builtin_amdgcn_s_wait_tensorcnt(2);
#else
    async_wait<4>();
#endif
    __syncthreads();                         // tile kt resident for all waves
    compute(kt & 1);
    __syncthreads();                         // reads done before re-staging
  }
#if USE_TDM
  if (wid == 0) __builtin_amdgcn_s_wait_tensorcnt(0);
#else
  async_wait<0>();
#endif
  __syncthreads();
  compute((nk - 1) & 1);

  #pragma unroll
  for (int i = 0; i < 2; ++i)
    #pragma unroll
    for (int j = 0; j < 2; ++j) {
      int mbase = m0 + wy * 32 + i * 16 + 8 * hi;
      int nn = n0 + wx * 32 + j * 16 + ln;
      #pragma unroll
      for (int r = 0; r < 8; ++r) {
        size_t o = (size_t)(mbase + r) * N + nn;
        float v = accv[i][j][r];
        if (acc) C[o] += v; else C[o] = v;
      }
    }
}

// ---------------------------------------------------------------------------
// Flash attention: 4 waves/block share one (b,h); each wave owns a 16-row
// q tile. K and V 32-key tiles async-staged into double-buffered LDS.
// Q/K/V: bf16 [b,h,l,64]. Output: bf16 [b, l, NH*64] (ready for out-proj).
// ---------------------------------------------------------------------------
__global__ __launch_bounds__(128) void attn_kernel(const u16* __restrict__ Qb,
                                                   const u16* __restrict__ Kb,
                                                   const u16* __restrict__ Vb,
                                                   u16* __restrict__ O,
                                                   int Lq, int Lk) {
  __shared__ alignas(16) u16 Klds[2][32 * 64];
  __shared__ alignas(16) u16 Vlds[2][32 * 64];
  __shared__ alignas(16) u16 Plds[4][16 * 32];

  int tid = threadIdx.x;
  int lane = tid & 31, wid = tid >> 5;
  int hi = lane >> 4, ln = lane & 15;
  int bh = blockIdx.y;
  int b = bh / NH, h = bh % NH;
  int qt = blockIdx.x * 4 + wid;

  // Q tile as two A fragments (K-dim = head dim, chunks of 32)
  const u16* qbase = Qb + ((size_t)bh * Lq + (size_t)qt * 16) * DH;
  Frag aq[2];
  #pragma unroll
  for (int c = 0; c < 2; ++c) {
    const u16* p = qbase + ln * DH + c * 32;
    aq[c].u[0] = *(const uint32x4*)(p + hi * 8);
    aq[c].u[1] = *(const uint32x4*)(p + 16 + hi * 8);
  }

  floatx8 oacc[4];
  float mrow[8], lrow[8];
  #pragma unroll
  for (int c = 0; c < 4; ++c) oacc[c] = (floatx8)0.0f;
  #pragma unroll
  for (int r = 0; r < 8; ++r) { mrow[r] = -3.0e38f; lrow[r] = 0.f; }

  // stage K & V 32x64 tiles into LDS buffer `buf` (4 async ops/thread)
  auto stage = [&](int buf, int kb) {
    const u16* kp = Kb + ((size_t)bh * Lk + (size_t)kb * 32) * DH;
    const u16* vp = Vb + ((size_t)bh * Lk + (size_t)kb * 32) * DH;
    #pragma unroll
    for (int i = 0; i < 2; ++i) {
      int c = (i * 128 + tid) * 8;
      async_copy16(&Klds[buf][c], kp + c);
      async_copy16(&Vlds[buf][c], vp + c);
    }
  };

  auto process = [&](int buf) {
    floatx8 s[2];
    #pragma unroll
    for (int n = 0; n < 2; ++n) {
      s[n] = (floatx8)0.0f;
      #pragma unroll
      for (int c = 0; c < 2; ++c) {
        // B = K^T : B^T rows are rows of the K matrix -> contiguous LDS reads
        Frag bk;
        const u16* pk = &Klds[buf][(n * 16 + ln) * DH + c * 32 + hi * 8];
        bk.u[0] = *(const uint32x4*)(pk);
        bk.u[1] = *(const uint32x4*)(pk + 16);
        s[n] = wmma_bf16(aq[c].v, bk.v, s[n]);
      }
    }

    // online softmax (row = r + 8*hi, spread over 16 lanes of this half)
    float alpha[8];
    #pragma unroll
    for (int r = 0; r < 8; ++r) {
      float mx = fmaxf(s[0][r], s[1][r]);
      mx = fmaxf(mx, __shfl_xor(mx, 1));
      mx = fmaxf(mx, __shfl_xor(mx, 2));
      mx = fmaxf(mx, __shfl_xor(mx, 4));
      mx = fmaxf(mx, __shfl_xor(mx, 8));
      float mn = fmaxf(mrow[r], mx);
      float al = __expf(mrow[r] - mn);
      mrow[r] = mn; alpha[r] = al;
      float p0 = __expf(s[0][r] - mn), p1 = __expf(s[1][r] - mn);
      s[0][r] = p0; s[1][r] = p1;
      float rs = p0 + p1;
      rs += __shfl_xor(rs, 1);
      rs += __shfl_xor(rs, 2);
      rs += __shfl_xor(rs, 4);
      rs += __shfl_xor(rs, 8);
      lrow[r] = lrow[r] * al + rs;
    }
    #pragma unroll
    for (int c = 0; c < 4; ++c)
      #pragma unroll
      for (int r = 0; r < 8; ++r) oacc[c][r] *= alpha[r];

    // P (C-layout) -> per-wave LDS region [16 x 32] bf16 (same-wave DS order)
    #pragma unroll
    for (int r = 0; r < 8; ++r) {
      int m = r + 8 * hi;
      Plds[wid][m * 32 + ln]      = f2bf(s[0][r]);
      Plds[wid][m * 32 + 16 + ln] = f2bf(s[1][r]);
    }

    // P as A fragment
    Frag ap;
    ap.u[0] = *(const uint32x4*)(&Plds[wid][ln * 32 + hi * 8]);
    ap.u[1] = *(const uint32x4*)(&Plds[wid][ln * 32 + 16 + hi * 8]);
    // V as B fragments (needs V^T -> gather from LDS)
    #pragma unroll
    for (int c = 0; c < 4; ++c) {
      Frag bv;
      #pragma unroll
      for (int j = 0; j < 16; ++j)
        bv.s[j] = Vlds[buf][(hi * 8 + j + (j >= 8 ? 8 : 0)) * DH + c * 16 + ln];
      oacc[c] = wmma_bf16(ap.v, bv.v, oacc[c]);
    }
  };

  int nkb = Lk / 32;
  stage(0, 0);
  for (int kbi = 0; kbi + 1 < nkb; ++kbi) {  // steady state: no conditionals
    stage((kbi + 1) & 1, kbi + 1);
    async_wait<4>();
    __syncthreads();                          // K/V tile kbi resident
    process(kbi & 1);
    __syncthreads();                          // reads done before re-staging
  }
  async_wait<0>();
  __syncthreads();
  process((nkb - 1) & 1);

  // write O / l  as bf16 into (b, l, h*64 + e)
  u16* ob = O + ((size_t)b * Lq + (size_t)qt * 16) * DM + h * DH;
  #pragma unroll
  for (int c = 0; c < 4; ++c)
    #pragma unroll
    for (int r = 0; r < 8; ++r) {
      int m = r + 8 * hi;
      ob[(size_t)m * DM + c * 16 + ln] = f2bf(oacc[c][r] / lrow[r]);
    }
}

// ---------------------------------------------------------------------------
// Host launcher
// ---------------------------------------------------------------------------
extern "C" void kernel_launch(void* const* d_in, const int* in_sizes, int n_in,
                              void* d_out, int out_size, void* d_ws, size_t ws_size,
                              hipStream_t stream) {
  (void)in_sizes; (void)n_in; (void)out_size; (void)ws_size;

  const float* x        = (const float*)d_in[0];
  const float* pos      = (const float*)d_in[1];
  const float* x_cross  = (const float*)d_in[2];
  const float* pos_cross= (const float*)d_in[3];
  const float* cond     = (const float*)d_in[4];
  const float* sa_norm_w= (const float*)d_in[5];
  const float* sa_qkv_w = (const float*)d_in[6];
  const float* sa_scale = (const float*)d_in[7];
  const float* sa_out_w = (const float*)d_in[8];
  const float* ca_norm_w= (const float*)d_in[9];
  const float* ca_ncs   = (const float*)d_in[10];
  const float* ca_q_w   = (const float*)d_in[11];
  const float* ca_kv_w  = (const float*)d_in[12];
  const float* ca_scale = (const float*)d_in[13];
  const float* ca_out_w = (const float*)d_in[14];
  const float* ff_norm_w= (const float*)d_in[15];
  const float* ff_up_w  = (const float*)d_in[16];
  const float* ff_down_w= (const float*)d_in[17];
  float* out = (float*)d_out;

  // ---- workspace bump allocator (phase-reused regions) ----
  char* base = (char*)d_ws;
  size_t off = 0;
  auto alloc = [&](size_t bytes) -> void* {
    void* p = base + off;
    off += (bytes + 255) & ~(size_t)255;
    return p;
  };
  float* svec_sa = (float*)alloc((size_t)BB * DM * 4);
  float* svec_ca = (float*)alloc((size_t)BB * DM * 4);
  float* svec_ff = (float*)alloc((size_t)BB * DM * 4);
  u16* Wqkv = (u16*)alloc((size_t)3 * DM * DM * 2);
  u16* Wso  = (u16*)alloc((size_t)DM * DM * 2);
  u16* Wcq  = (u16*)alloc((size_t)DM * DM * 2);
  u16* Wckv = (u16*)alloc((size_t)2 * DM * DC * 2);
  u16* Wco  = (u16*)alloc((size_t)DM * DM * 2);
  u16* Wup  = (u16*)alloc((size_t)2 * DFF * DM * 2);
  u16* Wdn  = (u16*)alloc((size_t)DM * DFF * 2);
  u16* hbf  = (u16*)alloc((size_t)BB * LL * DM * 2);
  u16* hcbf = (u16*)alloc((size_t)BB * LLC * DC * 2);
  u16* attnO= (u16*)alloc((size_t)BB * LL * DM * 2);
  // 24MB region: SA/CA q,k,v bf16; later reused for FF gate output t
  u16* qkvb = (u16*)alloc((size_t)3 * BB * NH * LL * DH * 2);
  u16* qb = qkvb;
  u16* kb = qkvb + (size_t)BB * NH * LL * DH;
  u16* vb = kb  + (size_t)BB * NH * LL * DH;
  u16* tbf = qkvb;
  // 96MB fp32 region: qkv proj (48MB) -> CA q/kv -> FF up (96MB)
  float* bigf = (float*)alloc((size_t)BB * LL * 2 * DFF * 4);
  float* qkvf = bigf;
  float* qf   = bigf;
  float* kvf  = bigf + (size_t)BB * LL * DM;
  float* uf   = bigf;

  const int M  = BB * LL;    // 4096
  const int Mc = BB * LLC;   // 1024

  // residual stream lives in d_out
  copy_f32_kernel<<<(M * DM) / 256, 256, 0, stream>>>(x, out, M * DM);

  // cond projections (+1)
  cond_proj_kernel<<<(BB * DM) / 256, 256, 0, stream>>>(cond, sa_norm_w, svec_sa, DM);
  cond_proj_kernel<<<(BB * DM) / 256, 256, 0, stream>>>(cond, ca_norm_w, svec_ca, DM);
  cond_proj_kernel<<<(BB * DM) / 256, 256, 0, stream>>>(cond, ff_norm_w, svec_ff, DM);

  // weights -> bf16
  cvt_bf16_kernel<<<(3*DM*DM)/256, 256, 0, stream>>>(sa_qkv_w, Wqkv, 3*DM*DM);
  cvt_bf16_kernel<<<(DM*DM)/256,   256, 0, stream>>>(sa_out_w, Wso, DM*DM);
  cvt_bf16_kernel<<<(DM*DM)/256,   256, 0, stream>>>(ca_q_w,   Wcq, DM*DM);
  cvt_bf16_kernel<<<(2*DM*DC)/256, 256, 0, stream>>>(ca_kv_w,  Wckv, 2*DM*DC);
  cvt_bf16_kernel<<<(DM*DM)/256,   256, 0, stream>>>(ca_out_w, Wco, DM*DM);
  cvt_bf16_kernel<<<(2*DFF*DM)/256,256, 0, stream>>>(ff_up_w,  Wup, 2*DFF*DM);
  cvt_bf16_kernel<<<(DM*DFF)/256,  256, 0, stream>>>(ff_down_w,Wdn, DM*DFF);

  // ============ Self attention ============
  ada_rms_kernel<<<M, 256, 0, stream>>>(out, svec_sa, hbf, DM, 1, LL);
  gemm_bf16_kernel<<<dim3((3*DM)/64, M/64), 128, 0, stream>>>(hbf, Wqkv, qkvf, M, 3*DM, DM, 0);
  prep_qk_kernel<<<(BB*NH*LL)/4, 128, 0, stream>>>(qkvf, 3*DM, 0,    pos, sa_scale, qb, LL);
  prep_qk_kernel<<<(BB*NH*LL)/4, 128, 0, stream>>>(qkvf, 3*DM, DM,   pos, sa_scale, kb, LL);
  prep_v_kernel<<<(BB*NH*LL*DH)/256, 256, 0, stream>>>(qkvf, 3*DM, 2*DM, vb, LL, BB*NH*LL*DH);
  attn_kernel<<<dim3(LL/64, BB*NH), 128, 0, stream>>>(qb, kb, vb, attnO, LL, LL);
  gemm_bf16_kernel<<<dim3(DM/64, M/64), 128, 0, stream>>>(attnO, Wso, out, M, DM, DM, 1);

  // ============ Cross attention ============
  ada_rms_kernel<<<M, 256, 0, stream>>>(out, svec_ca, hbf, DM, 1, LL);
  ada_rms_kernel<<<Mc, 256, 0, stream>>>(x_cross, ca_ncs, hcbf, DC, 0, LLC);
  gemm_bf16_kernel<<<dim3(DM/64, M/64), 128, 0, stream>>>(hbf, Wcq, qf, M, DM, DM, 0);
  gemm_bf16_kernel<<<dim3((2*DM)/64, Mc/64), 128, 0, stream>>>(hcbf, Wckv, kvf, Mc, 2*DM, DC, 0);
  prep_qk_kernel<<<(BB*NH*LL)/4, 128, 0, stream>>>(qf, DM, 0, pos, ca_scale, qb, LL);
  prep_qk_kernel<<<(BB*NH*LLC)/4, 128, 0, stream>>>(kvf, 2*DM, 0, pos_cross, ca_scale, kb, LLC);
  prep_v_kernel<<<(BB*NH*LLC*DH)/256, 256, 0, stream>>>(kvf, 2*DM, DM, vb, LLC, BB*NH*LLC*DH);
  attn_kernel<<<dim3(LL/64, BB*NH), 128, 0, stream>>>(qb, kb, vb, attnO, LL, LLC);
  gemm_bf16_kernel<<<dim3(DM/64, M/64), 128, 0, stream>>>(attnO, Wco, out, M, DM, DM, 1);

  // ============ FFN (SwiGLU) ============
  ada_rms_kernel<<<M, 256, 0, stream>>>(out, svec_ff, hbf, DM, 1, LL);
  gemm_bf16_kernel<<<dim3((2*DFF)/64, M/64), 128, 0, stream>>>(hbf, Wup, uf, M, 2*DFF, DM, 0);
  silu_mul_kernel<<<(M*DFF)/256, 256, 0, stream>>>(uf, tbf, M*DFF);
  gemm_bf16_kernel<<<dim3(DM/64, M/64), 128, 0, stream>>>(tbf, Wdn, out, M, DM, DFF, 1);
}
